// MultiLayerPerceptron_he_36412732735948
// MI455X (gfx1250) — matrix operand order repaired and checked by reading, not compiled, not run on hardware
//
#include <hip/hip_runtime.h>
#include <hip/hip_bf16.h>
#include <stdint.h>

typedef _Float16 half_t;
typedef __attribute__((ext_vector_type(16))) _Float16 v16h;
typedef __attribute__((ext_vector_type(8)))  _Float16 v8h;
typedef __attribute__((ext_vector_type(8)))  float    v8f;

#define BATCH 32
#define NPTS  1024
#define FEAT  7
#define HID   256
#define MLPH  64
#define NFLAT (NPTS * FEAT)   /* 7168 */
#define THR2  0.09f           /* 0.3^2 */

/* ---------------- workspace layout (bytes, all 256B-aligned) ------------- */
static constexpr size_t OFF_DINV  = 0;                         /* [B,N] f32   131072 */
static constexpr size_t OFF_SPEED = OFF_DINV  + 131072;        /* [B]   f32      256 */
static constexpr size_t OFF_MAXD2 = OFF_SPEED + 256;           /* [B]   f32      256 */
static constexpr size_t OFF_HSUM  = OFF_MAXD2 + 256;           /* [B,256] f32  32768 */
static constexpr size_t OFF_UH    = OFF_HSUM  + 32768;         /* [B,256] f16  16384 */
static constexpr size_t OFF_XFH   = OFF_UH    + 16384;         /* [32,7168] f16 458752 */
static constexpr size_t OFF_WM0H  = OFF_XFH   + 458752;        /* [64,7168] f16 917504 */
static constexpr size_t OFF_WM1H  = OFF_WM0H  + 917504;        /* [64,64]  f16   8192 */
static constexpr size_t OFF_W2H   = OFF_WM1H  + 8192;          /* [256,256] f16 131072 */
static constexpr size_t OFF_WFCH  = OFF_W2H   + 131072;        /* [256,256] f16 131072 */
static constexpr size_t OFF_M0F   = OFF_WFCH  + 131072;        /* [32,64] f32    8192 */
static constexpr size_t OFF_M0H   = OFF_M0F   + 8192;          /* [32,64] f16    4096 */
static constexpr size_t OFF_M1F   = OFF_M0H   + 4096;          /* [32,64] f32    8192 */
static constexpr size_t OFF_M1H   = OFF_M1F   + 8192;          /* [32,64] f16    4096 */
static constexpr size_t OFF_AF    = OFF_M1H   + 4096;          /* [32,256] f32  32768 */
static constexpr size_t OFF_AH    = OFF_AF    + 32768;         /* [32,256] f16  16384 */
static constexpr size_t OFF_GCNF  = OFF_AH    + 16384;         /* [32,256] f32  32768 */
static constexpr size_t OFF_GCNH  = OFF_GCNF  + 32768;         /* [32,256] f16  16384 */

/* ------------------------------ conversion ------------------------------- */
__global__ void cast_to_half_kernel(const float* __restrict__ src,
                                    half_t* __restrict__ dst, int n) {
  for (int i = blockIdx.x * blockDim.x + threadIdx.x; i < n;
       i += gridDim.x * blockDim.x)
    dst[i] = (half_t)src[i];
}

/* ---------------- pass 1: degree, max pairwise d^2, mean speed ----------- */
__global__ __launch_bounds__(256) void pairwise1_kernel(
    const float* __restrict__ x, float* __restrict__ dinv,
    float* __restrict__ speedsum, float* __restrict__ maxd2) {
  __shared__ float px[NPTS], py[NPTS], red[256];
  const int b = blockIdx.x, tid = threadIdx.x;
  const float* xb = x + (size_t)b * NPTS * FEAT;

  float sp = 0.0f;
  for (int j = tid; j < NPTS; j += 256) {
    px[j] = xb[j * FEAT + 1];
    py[j] = xb[j * FEAT + 2];
    float vx = xb[j * FEAT + 3], vy = xb[j * FEAT + 4];
    sp += sqrtf(vx * vx + vy * vy);
  }
  __syncthreads();

  float mloc = 0.0f;
  for (int r = 0; r < 4; ++r) {
    int i = r * 256 + tid;
    float xi = px[i], yi = py[i];
    int deg = 0;
    float m = 0.0f;
    for (int j = 0; j < NPTS; ++j) {
      float dx = xi - px[j], dy = yi - py[j];
      float d2 = dx * dx + dy * dy;
      deg += (d2 < THR2) ? 1 : 0;
      m = fmaxf(m, d2);
    }
    dinv[b * NPTS + i] = 1.0f / sqrtf((float)deg);  /* deg >= 1 (self loop) */
    mloc = fmaxf(mloc, m);
  }
  red[tid] = mloc;
  __syncthreads();
  for (int s = 128; s > 0; s >>= 1) {
    if (tid < s) red[tid] = fmaxf(red[tid], red[tid + s]);
    __syncthreads();
  }
  if (tid == 0) maxd2[b] = red[0];
  __syncthreads();
  red[tid] = sp;
  __syncthreads();
  for (int s = 128; s > 0; s >>= 1) {
    if (tid < s) red[tid] += red[tid + s];
    __syncthreads();
  }
  if (tid == 0) speedsum[b] = red[0];
}

/* ------- pass 2: s_i, c_i; fused h=relu(W1 s + b1); weighted colsum ------ */
/* hsum[b,k] = sum_i (c_i*dinv_i/N) * relu(W1[k,:]·s_i + b1[k])             */
__global__ __launch_bounds__(256) void pairwise2_kernel(
    const float* __restrict__ x, const float* __restrict__ dinv,
    const float* __restrict__ W1, const float* __restrict__ b1,
    float* __restrict__ hsum) {
  __shared__ float px[NPTS], py[NPTS], dv[NPTS];
  __shared__ float ss0[NPTS], ss1[NPTS], sw[NPTS];
  __shared__ float w1l[2 * HID], b1l[HID];
  const int b = blockIdx.x, tid = threadIdx.x;
  const float* xb = x + (size_t)b * NPTS * FEAT;

  for (int j = tid; j < NPTS; j += 256) {
    px[j] = xb[j * FEAT + 1];
    py[j] = xb[j * FEAT + 2];
    dv[j] = dinv[b * NPTS + j];
  }
  for (int k = tid; k < 2 * HID; k += 256) w1l[k] = W1[k];
  b1l[tid] = b1[tid];
  __syncthreads();

  float rs0[4], rs1[4], rw[4];
  for (int r = 0; r < 4; ++r) {
    int i = r * 256 + tid;
    float xi = px[i], yi = py[i], di = dv[i];
    float a0 = 0.0f, a1 = 0.0f, cc = 0.0f;
    for (int j = 0; j < NPTS; ++j) {
      float dx = xi - px[j], dy = yi - py[j];
      float d2 = dx * dx + dy * dy;
      float w = (d2 < THR2) ? dv[j] : 0.0f;
      cc += w;
      a0 += w * px[j];
      a1 += w * py[j];
    }
    rs0[r] = a0 * di;
    rs1[r] = a1 * di;
    rw[r]  = cc * di * (1.0f / (float)NPTS);
  }
  __syncthreads();
  for (int r = 0; r < 4; ++r) {
    int i = r * 256 + tid;
    ss0[i] = rs0[r]; ss1[i] = rs1[r]; sw[i] = rw[r];
  }
  __syncthreads();
  /* each thread owns one hidden channel -> deterministic accumulation */
  {
    int k = tid;
    float w0 = w1l[2 * k], w1v = w1l[2 * k + 1], bb = b1l[k];
    float acc = 0.0f;
    for (int i = 0; i < NPTS; ++i)
      acc += sw[i] * fmaxf(w0 * ss0[i] + w1v * ss1[i] + bb, 0.0f);
    hsum[b * HID + k] = acc;
  }
}

__global__ void finalize_u_kernel(const float* __restrict__ hsum,
                                  half_t* __restrict__ u_h) {
  int i = blockIdx.x * 256 + threadIdx.x;   /* grid = BATCH, so i < B*HID */
  u_h[i] = (half_t)hsum[i];                 /* already includes 1/N factor */
}

/* ---------- generic WMMA GEMM: C = A[MxK] * W^T + bias (W:[N,K]) --------- */
/* A f16 row-major [M,K]; W f16 row-major [Nout,K] (native torch layout).   */
/* One wave per 16x16 output tile; f32 accumulate.                          */
/* B-fragment trick: B[k][n] = W[n][k], so, in table order, each    */
/* lane takes column n as two 16B chunks of W row n: K 8h+0..7, 16+8h+0..7.   */
__global__ __launch_bounds__(32) void wmma_gemm_kernel(
    const half_t* __restrict__ A, const half_t* __restrict__ W,
    const float* __restrict__ bias, float* __restrict__ Cf,
    half_t* __restrict__ Ch, int M, int Nn, int K, int do_relu) {
  const int m0 = blockIdx.x * 16;
  const int n0 = blockIdx.y * 16;
  const int lane = threadIdx.x;
  const int hid  = lane >> 4;   /* 0: lanes 0-15, 1: lanes 16-31 */
  const int lr   = lane & 15;

  const float bv = bias[n0 + lr];
  v8f c;
#pragma unroll
  for (int i = 0; i < 8; ++i) c[i] = bv;

  const half_t* arow = A + (size_t)(m0 + lr) * K + hid * 8;
  const half_t* wrow = W + (size_t)(n0 + lr) * K + hid * 8;

  for (int k0 = 0; k0 < K; k0 += 32) {
    /* A frag: lanes 0-15 -> K {0..7,16..23}; lanes 16-31 -> {8..15,24..31} */
    v8h lo = *(const v8h*)(arow + k0);
    v8h hi = *(const v8h*)(arow + k0 + 16);
    v16h a;
#pragma unroll
    for (int i = 0; i < 8; ++i) { a[i] = lo[i]; a[8 + i] = hi[i]; }

    /* B frag: col n0+lr; lanes 0-15 -> K {0..7,16..23}; lanes 16-31 -> {8..15,24..31} */
    v8h blo = *(const v8h*)(wrow + k0);
    v8h bhi = *(const v8h*)(wrow + k0 + 16);
    v16h bfr;
#pragma unroll
    for (int i = 0; i < 8; ++i) { bfr[i] = blo[i]; bfr[8 + i] = bhi[i]; }

    c = __builtin_amdgcn_wmma_f32_16x16x32_f16(
        /*neg_a=*/false, a, /*neg_b=*/false, bfr,
        /*c_mod=*/(short)0, c, /*reuse_a=*/false, /*reuse_b=*/false);
  }

#pragma unroll
  for (int vk = 0; vk < 8; ++vk) {
    float val = c[vk];
    if (do_relu) val = fmaxf(val, 0.0f);
    int row = m0 + hid * 8 + vk;
    int col = n0 + lr;
    Cf[(size_t)row * Nn + col] = val;
    Ch[(size_t)row * Nn + col] = (half_t)val;
  }
}

/* --------------- final: concat [mlp(64) | gcn(256) | glo(8)] @ Wp^T ------ */
__global__ __launch_bounds__(256) void final_kernel(
    const float* __restrict__ m1f, const float* __restrict__ gcnf,
    const float* __restrict__ speedsum, const float* __restrict__ maxd2,
    const float* __restrict__ Wg, const float* __restrict__ bg,
    const float* __restrict__ Wp, const float* __restrict__ bp,
    float* __restrict__ out) {
  const int t = threadIdx.x;           /* 0..255 -> (b,o) */
  const int b = t >> 3, o = t & 7;
  const float speed = speedsum[b] * (1.0f / (float)NPTS);
  const float dens  = 1.0f / sqrtf(maxd2[b]);
  const float* wrow = Wp + o * (MLPH + HID + 8);
  float acc = bp[o];
  for (int k = 0; k < MLPH; ++k) acc += m1f[b * MLPH + k] * wrow[k];
  for (int k = 0; k < HID; ++k)  acc += gcnf[b * HID + k] * wrow[MLPH + k];
  for (int j = 0; j < 8; ++j) {
    float g = fmaxf(Wg[j * 2] * speed + Wg[j * 2 + 1] * dens + bg[j], 0.0f);
    acc += g * wrow[MLPH + HID + j];
  }
  out[t] = acc;
}

/* -------------------------------- launch --------------------------------- */
extern "C" void kernel_launch(void* const* d_in, const int* in_sizes, int n_in,
                              void* d_out, int out_size, void* d_ws,
                              size_t ws_size, hipStream_t stream) {
  (void)in_sizes; (void)n_in; (void)out_size; (void)ws_size;
  const float* x   = (const float*)d_in[0];
  const float* W1  = (const float*)d_in[1];
  const float* b1  = (const float*)d_in[2];
  const float* W2  = (const float*)d_in[3];
  const float* b2  = (const float*)d_in[4];
  const float* Wfc = (const float*)d_in[5];
  const float* bfc = (const float*)d_in[6];
  const float* Wg  = (const float*)d_in[7];
  const float* bg  = (const float*)d_in[8];
  const float* Wm0 = (const float*)d_in[9];
  const float* bm0 = (const float*)d_in[10];
  const float* Wm1 = (const float*)d_in[11];
  const float* bm1 = (const float*)d_in[12];
  const float* Wp  = (const float*)d_in[13];
  const float* bp  = (const float*)d_in[14];
  float* out = (float*)d_out;

  uint8_t* ws = (uint8_t*)d_ws;
  float*  dinv     = (float*)(ws + OFF_DINV);
  float*  speedsum = (float*)(ws + OFF_SPEED);
  float*  maxd2    = (float*)(ws + OFF_MAXD2);
  float*  hsum     = (float*)(ws + OFF_HSUM);
  half_t* uh       = (half_t*)(ws + OFF_UH);
  half_t* xfh      = (half_t*)(ws + OFF_XFH);
  half_t* wm0h     = (half_t*)(ws + OFF_WM0H);
  half_t* wm1h     = (half_t*)(ws + OFF_WM1H);
  half_t* w2h      = (half_t*)(ws + OFF_W2H);
  half_t* wfch     = (half_t*)(ws + OFF_WFCH);
  float*  m0f      = (float*)(ws + OFF_M0F);
  half_t* m0h      = (half_t*)(ws + OFF_M0H);
  float*  m1f      = (float*)(ws + OFF_M1F);
  half_t* m1h      = (half_t*)(ws + OFF_M1H);
  float*  af       = (float*)(ws + OFF_AF);
  half_t* ah       = (half_t*)(ws + OFF_AH);
  float*  gcnf     = (float*)(ws + OFF_GCNF);
  half_t* gcnh     = (half_t*)(ws + OFF_GCNH);

  /* casts to f16 (weights stay in native [Nout,K] layout) */
  cast_to_half_kernel<<<256, 256, 0, stream>>>(x, xfh, BATCH * NFLAT);
  cast_to_half_kernel<<<256, 256, 0, stream>>>(Wm0, wm0h, MLPH * NFLAT);
  cast_to_half_kernel<<<16, 256, 0, stream>>>(Wm1, wm1h, MLPH * MLPH);
  cast_to_half_kernel<<<64, 256, 0, stream>>>(W2, w2h, HID * HID);
  cast_to_half_kernel<<<64, 256, 0, stream>>>(Wfc, wfch, HID * HID);

  /* GCN pairwise passes (one workgroup per batch, all data in LDS) */
  pairwise1_kernel<<<BATCH, 256, 0, stream>>>(x, dinv, speedsum, maxd2);
  pairwise2_kernel<<<BATCH, 256, 0, stream>>>(x, dinv, W1, b1, hsum);
  finalize_u_kernel<<<BATCH, 256, 0, stream>>>(hsum, uh);

  /* MLP branch: [32,7168]@Wm0^T+relu ; [32,64]@Wm1^T+relu */
  wmma_gemm_kernel<<<dim3(2, 4), 32, 0, stream>>>(xfh, wm0h, bm0, m0f, m0h,
                                                  32, MLPH, NFLAT, 1);
  wmma_gemm_kernel<<<dim3(2, 4), 32, 0, stream>>>(m0h, wm1h, bm1, m1f, m1h,
                                                  32, MLPH, MLPH, 1);
  /* GCN tail: a = u@W2^T+b2 ; gcn_en = a@Wfc^T+bfc */
  wmma_gemm_kernel<<<dim3(2, 16), 32, 0, stream>>>(uh, w2h, b2, af, ah,
                                                   32, HID, HID, 0);
  wmma_gemm_kernel<<<dim3(2, 16), 32, 0, stream>>>(ah, wfch, bfc, gcnf, gcnh,
                                                   32, HID, HID, 0);

  /* concat + projection + global branch */
  final_kernel<<<1, 256, 0, stream>>>(m1f, gcnf, speedsum, maxd2,
                                      Wg, bg, Wp, bp, out);
}
